// DualAttnTransformerLM_52939766891023
// MI455X (gfx1250) — matrix operand order repaired and checked by reading, not compiled, not run on hardware
//
#include <hip/hip_runtime.h>

// ---------------------------------------------------------------------------
// Types / constants
// ---------------------------------------------------------------------------
typedef __bf16 bf16_t;
typedef __attribute__((ext_vector_type(16))) __bf16 v16bf;
typedef __attribute__((ext_vector_type(8)))  float  v8f;
typedef int v4i_raw __attribute__((vector_size(16)));   // GCC vector, matches builtin

#define L_SEQ   2048
#define DMODEL  1024
#define NHEAD   8
#define KDIM    64
#define HDIM    64
#define RDIM    128
#define NPROJ   (NHEAD * HDIM)      // 512
#define ATTN_SCALE 0.125f           // 1/sqrt(64)
#define REL_SCALE  0.08838834764831845f  // 1/sqrt(128)

// CDNA5 async global->LDS path (guarded: falls back to plain loads if the
// toolchain does not declare the builtins).
#if defined(__has_builtin)
#if __has_builtin(__builtin_amdgcn_global_load_async_to_lds_b128) && \
    __has_builtin(__builtin_amdgcn_s_wait_asynccnt)
#define HAVE_ASYNC_LDS 1
#endif
#endif

union FragU {
    v16bf v;
    int4  q[2];
};

__device__ __forceinline__ v8f zero8() {
    v8f z;
#pragma unroll
    for (int i = 0; i < 8; ++i) z[i] = 0.0f;
    return z;
}

__device__ __forceinline__ v8f wmma_bf16(v16bf a, v16bf b, v8f c) {
    // D = A(16x32 bf16) * B(32x16 bf16) + C(16x16 f32)
    return __builtin_amdgcn_wmma_f32_16x16x32_bf16(
        /*neg_a=*/false, a, /*neg_b=*/false, b,
        /*c_mod=*/(short)0, c, /*reuse_a=*/false, /*reuse_b=*/false);
}

// A-operand 16x32 bf16 tile at `base` (row-major, row stride = `stride` elems).
// ISA layout: lanes 0-15 = M, K chunks {0..7,16..23}; lanes 16-31 K {8..15,24..31}.
__device__ __forceinline__ v16bf load_frag_a(const bf16_t* base, int stride, int lane) {
    int m  = lane & 15;
    int k0 = (lane >> 4) * 8;
    const bf16_t* p = base + (long)m * stride + k0;
    FragU f;
    f.q[0] = *reinterpret_cast<const int4*>(p);
    f.q[1] = *reinterpret_cast<const int4*>(p + 16);
    return f.v;
}

// B-operand 32x16 bf16 tile, stored transposed (N rows x K cols, row-major).
// Lane n (0-15) holds K chunks {0..7,16..23}; lanes 16-31 K {8..15,24..31}.
__device__ __forceinline__ v16bf load_frag_b(const bf16_t* baseT, int stride, int lane) {
    int n  = lane & 15;
    int k0 = (lane >> 4) * 8;
    const bf16_t* p = baseT + (long)n * stride + k0;
    FragU f;
    f.q[0] = *reinterpret_cast<const int4*>(p);
    f.q[1] = *reinterpret_cast<const int4*>(p + 16);
    return f.v;
}

// ---------------------------------------------------------------------------
// Kernel 1: fp32 -> bf16 convert (optionally transposing)
// ---------------------------------------------------------------------------
__global__ void convert_kernel(const float* __restrict__ src, bf16_t* __restrict__ dst,
                               int rows, int cols, int do_transpose) {
    int n = rows * cols;
    for (int idx = blockIdx.x * blockDim.x + threadIdx.x; idx < n;
         idx += gridDim.x * blockDim.x) {
        int r = idx / cols;
        int c = idx - r * cols;
        float v = src[idx];
        if (do_transpose) dst[(long)c * rows + r] = (bf16_t)v;
        else              dst[idx] = (bf16_t)v;
    }
}

// ---------------------------------------------------------------------------
// Kernel 2: bf16 WMMA GEMM, C = A(MxK) * Bt^T (Bt stored N x K).
// One wave computes a 16(M) x 64(N) tile: per K-step, 1 A-frag is reused by
// 4 B-frags -> 4 independent WMMA accumulator chains.  Epilogue modes:
//   1: fused RoPE, scatter bf16 to (B, NHEAD, L, KDIM)
//   2: scatter bf16 transposed to (B, NHEAD, HDIM, L)  (svT)
//   3: scatter bf16 to (B, NREL=8, L, RDIM)
//   4: fp32 row-major store (final output)
// ---------------------------------------------------------------------------
__global__ void gemm_bf16_kernel(const bf16_t* __restrict__ A,
                                 const bf16_t* __restrict__ Bt,
                                 void* __restrict__ Cout,
                                 int M, int N, int K, int mode,
                                 const float* __restrict__ cosT,
                                 const float* __restrict__ sinT) {
    int lane = threadIdx.x & 31;
    int m0 = blockIdx.x * 16;
    int n0 = blockIdx.y * 64;

    const bf16_t* ap = A  + (long)m0 * K;
    const bf16_t* bp = Bt + (long)n0 * K;

    v8f acc[4];
#pragma unroll
    for (int nt = 0; nt < 4; ++nt) acc[nt] = zero8();

    for (int k0 = 0; k0 < K; k0 += 32) {
        __builtin_prefetch(ap + k0 + 128, 0, 1);   // global_prefetch_b8
        __builtin_prefetch(bp + k0 + 128, 0, 1);
        v16bf a = load_frag_a(ap + k0, K, lane);
#pragma unroll
        for (int nt = 0; nt < 4; ++nt) {
            v16bf b = load_frag_b(bp + (long)nt * 16 * K + k0, K, lane);
            acc[nt] = wmma_bf16(a, b, acc[nt]);
        }
    }

    int colN    = lane & 15;
    int rowbase = (lane >> 4) * 8;

#pragma unroll
    for (int nt = 0; nt < 4; ++nt) {
        int col = n0 + nt * 16 + colN;

        if (mode == 1) {                   // RoPE + scatter to (B,8,L,64)
            bf16_t* dst = (bf16_t*)Cout;
            int t = (col & 63) >> 1;       // rotary freq index within the head
            int h = col >> 6;
            int kd = col & 63;
            bool odd = (col & 1) != 0;
#pragma unroll
            for (int v = 0; v < 8; ++v) {
                int row = m0 + rowbase + v;  // row = b*L + l
                int l = row & (L_SEQ - 1);
                int b = row >> 11;
                float c = acc[nt][v];
                float o = __shfl_xor(c, 1);  // partner of the rotary pair
                float cs = cosT[l * (KDIM / 2) + t];
                float sn = sinT[l * (KDIM / 2) + t];
                // even col: tr*cos - ti*sin ; odd col: tr*sin + ti*cos
                float res = odd ? (o * sn + c * cs) : (c * cs - o * sn);
                dst[(((long)b * NHEAD + h) * L_SEQ + l) * KDIM + kd] = (bf16_t)res;
            }
        } else if (mode == 2) {            // svT: (B,8,64,L)
            bf16_t* dst = (bf16_t*)Cout;
            int h = col >> 6;
            int hd = col & 63;
#pragma unroll
            for (int v = 0; v < 8; ++v) {
                int row = m0 + rowbase + v;
                int l = row & (L_SEQ - 1);
                int b = row >> 11;
                dst[(((long)b * NHEAD + h) * HDIM + hd) * L_SEQ + l] = (bf16_t)acc[nt][v];
            }
        } else if (mode == 3) {            // qr/kr: (B,8,L,128)
            bf16_t* dst = (bf16_t*)Cout;
            int r = col >> 7;
            int rd = col & 127;
#pragma unroll
            for (int v = 0; v < 8; ++v) {
                int row = m0 + rowbase + v;
                int l = row & (L_SEQ - 1);
                int b = row >> 11;
                dst[(((long)b * NHEAD + r) * L_SEQ + l) * RDIM + rd] = (bf16_t)acc[nt][v];
            }
        } else {                           // mode 4: fp32 row-major
            float* dst = (float*)Cout;
#pragma unroll
            for (int v = 0; v < 8; ++v) {
                int row = m0 + rowbase + v;
                dst[(long)row * N + col] = acc[nt][v];
            }
        }
    }
}

// ---------------------------------------------------------------------------
// Kernel 3: fused flash attention + relational attention.
// One workgroup = (b, 16-row i-tile); 8 waves, wave w = head h = rel-head r.
// ---------------------------------------------------------------------------
__global__ __launch_bounds__(256) void attn_kernel(
    const bf16_t* __restrict__ qh,   // (B,8,L,64)   roped
    const bf16_t* __restrict__ kh,   // (B,8,L,64)   roped
    const bf16_t* __restrict__ qrb,  // (B,8,L,128)
    const bf16_t* __restrict__ krb,  // (B,8,L,128)
    const bf16_t* __restrict__ svT,  // (B,8,64,L)
    const float*  __restrict__ wr,   // (8, 512) fp32
    bf16_t* __restrict__ Y)          // (B*L, 512) bf16
{
    __shared__ float rel_lds[16][32][8];                  // rel[i_loc][j_loc][r]
    __shared__ __align__(16) bf16_t p_lds[NHEAD][16][32]; // per-wave P staging
    __shared__ __align__(16) float wr_lds[8][NPROJ];      // wr native layout [r][h*64+d]

    int lane = threadIdx.x & 31;
    int w    = threadIdx.x >> 5;        // wave index = head h = rel head r
    int b    = blockIdx.x >> 7;         // 128 i-tiles per batch
    int i0   = (blockIdx.x & 127) << 4;

    // stage wr (16 KB) into LDS as a straight copy
#if HAVE_ASYNC_LDS
    {
        float* wl = &wr_lds[0][0];
        typedef __attribute__((address_space(1))) v4i_raw gv4i_t;
        typedef __attribute__((address_space(3))) v4i_raw lv4i_t;
#pragma unroll
        for (int it = 0; it < 4; ++it) {
            int e = (it * 256 + (int)threadIdx.x) * 4;   // 16B chunk per thread
            gv4i_t* gp = (gv4i_t*)(unsigned long long)(wr + e);
            lv4i_t* lp = (lv4i_t*)(unsigned)(unsigned long long)(wl + e);
            __builtin_amdgcn_global_load_async_to_lds_b128(gp, lp, 0, 0);
        }
        __builtin_amdgcn_s_wait_asynccnt(0);
    }
#else
    for (int idx = threadIdx.x; idx < 8 * NPROJ; idx += 256) {
        (&wr_lds[0][0])[idx] = wr[idx];
    }
#endif

    int colN    = lane & 15;
    int rowbase = (lane >> 4) * 8;

    // resident A-fragments: q (16x64) and qr (16x128) for this head
    const bf16_t* qbase  = qh  + (((long)b * NHEAD + w) * L_SEQ + i0) * KDIM;
    const bf16_t* qrbase = qrb + (((long)b * NHEAD + w) * L_SEQ + i0) * RDIM;
    v16bf qA0 = load_frag_a(qbase, KDIM, lane);
    v16bf qA1 = load_frag_a(qbase + 32, KDIM, lane);
    v16bf qrA[4];
#pragma unroll
    for (int kk = 0; kk < 4; ++kk) qrA[kk] = load_frag_a(qrbase + 32 * kk, RDIM, lane);

    const bf16_t* kbase  = kh  + (((long)b * NHEAD + w) * L_SEQ) * KDIM;
    const bf16_t* krbase = krb + (((long)b * NHEAD + w) * L_SEQ) * RDIM;
    const bf16_t* svbase = svT + (((long)b * NHEAD + w) * HDIM) * L_SEQ;

    v8f accv[4];                         // att_sym accumulator, N = 64
#pragma unroll
    for (int nd = 0; nd < 4; ++nd) accv[nd] = zero8();
    float t_acc[8][8];                   // per-lane partial t_rel[i][r]
#pragma unroll
    for (int v = 0; v < 8; ++v)
#pragma unroll
        for (int r = 0; r < 8; ++r) t_acc[v][r] = 0.0f;
    float m_st[8], l_st[8];
#pragma unroll
    for (int v = 0; v < 8; ++v) { m_st[v] = -1.0e30f; l_st[v] = 0.0f; }

    __syncthreads();                     // wr_lds ready (all waves' DMAs done)

    int jend = i0 + 16;                  // causal bound (exclusive)
    for (int j0 = 0; j0 < jend; j0 += 32) {
        // ---- (a) wave w computes rel[:, :, r=w] for this 16x32 tile -------
#pragma unroll
        for (int js = 0; js < 2; ++js) {
            const bf16_t* krrow = krbase + (long)(j0 + js * 16) * RDIM;
            v8f c = zero8();
#pragma unroll
            for (int kk = 0; kk < 4; ++kk)
                c = wmma_bf16(qrA[kk], load_frag_b(krrow + 32 * kk, RDIM, lane), c);
#pragma unroll
            for (int v = 0; v < 8; ++v)
                rel_lds[rowbase + v][js * 16 + colN][w] = c[v] * REL_SCALE;
        }
        __syncthreads();                 // rel tile visible to all heads

        // ---- (b) attention scores S = q k^T -------------------------------
        float s[2][8];
#pragma unroll
        for (int js = 0; js < 2; ++js) {
            const bf16_t* krow = kbase + (long)(j0 + js * 16) * KDIM;
            v8f c = zero8();
            c = wmma_bf16(qA0, load_frag_b(krow, KDIM, lane), c);
            c = wmma_bf16(qA1, load_frag_b(krow + 32, KDIM, lane), c);
#pragma unroll
            for (int v = 0; v < 8; ++v) {
                int ig = i0 + rowbase + v;
                int jg = j0 + js * 16 + colN;
                s[js][v] = (jg <= ig) ? c[v] * ATTN_SCALE : -1.0e30f;
            }
        }

        // ---- (c) online softmax (row lives in 16 lanes of a half-wave) ----
#pragma unroll
        for (int v = 0; v < 8; ++v) {
            float rm = fmaxf(s[0][v], s[1][v]);
#pragma unroll
            for (int mk = 8; mk >= 1; mk >>= 1) rm = fmaxf(rm, __shfl_xor(rm, mk));
            float mnew  = fmaxf(m_st[v], rm);
            float alpha = __expf(m_st[v] - mnew);
            float p0 = __expf(s[0][v] - mnew);
            float p1 = __expf(s[1][v] - mnew);
            s[0][v] = p0; s[1][v] = p1;
            float rs = p0 + p1;
#pragma unroll
            for (int mk = 8; mk >= 1; mk >>= 1) rs += __shfl_xor(rs, mk);
            l_st[v] = l_st[v] * alpha + rs;
            m_st[v] = mnew;
#pragma unroll
            for (int nd = 0; nd < 4; ++nd) accv[nd][v] *= alpha;
#pragma unroll
            for (int r = 0; r < 8; ++r) t_acc[v][r] *= alpha;
        }

        // ---- (d,e) stage P to LDS (bf16) + per-row t_rel accumulation -----
#pragma unroll
        for (int js = 0; js < 2; ++js)
#pragma unroll
            for (int v = 0; v < 8; ++v) {
                float pv = s[js][v];
                p_lds[w][rowbase + v][js * 16 + colN] = (bf16_t)pv;
                const float* rl = rel_lds[rowbase + v][js * 16 + colN];
#pragma unroll
                for (int r = 0; r < 8; ++r) t_acc[v][r] += pv * rl[r];
            }

        // compiler barrier: p_lds bf16 stores must not be reordered past the
        // int4-typed fragment reload below (HW DS ops are in-order per wave).
        asm volatile("" ::: "memory");

        // ---- (f) att_sym += P @ svT  (K = 32 = whole j-tile) --------------
        v16bf pA = load_frag_a(&p_lds[w][0][0], 32, lane);
#pragma unroll
        for (int nd = 0; nd < 4; ++nd) {
            const bf16_t* svrow = svbase + (long)(nd * 16) * L_SEQ + j0;
            accv[nd] = wmma_bf16(pA, load_frag_b(svrow, L_SEQ, lane), accv[nd]);
        }
        __syncthreads();                 // rel_lds reused next iteration
    }

    // ---- epilogue ---------------------------------------------------------
    float inv_l[8];
#pragma unroll
    for (int v = 0; v < 8; ++v) inv_l[v] = 1.0f / l_st[v];

    // reduce t_rel across the 16 lanes holding each row
#pragma unroll
    for (int v = 0; v < 8; ++v)
#pragma unroll
        for (int r = 0; r < 8; ++r) {
            float t = t_acc[v][r];
#pragma unroll
            for (int mk = 8; mk >= 1; mk >>= 1) t += __shfl_xor(t, mk);
            t_acc[v][r] = t;             // full sum, replicated in half-wave
        }

#pragma unroll
    for (int nd = 0; nd < 4; ++nd)
#pragma unroll
        for (int v = 0; v < 8; ++v) {
            int d = nd * 16 + colN;
            float srel = 0.0f;
#pragma unroll
            for (int r = 0; r < 8; ++r) srel += t_acc[v][r] * wr_lds[r][w * HDIM + d];
            float val = (accv[nd][v] + srel) * inv_l[v];
            int ig = i0 + rowbase + v;
            Y[((long)b * L_SEQ + ig) * NPROJ + w * HDIM + d] = (bf16_t)val;
        }
}

// ---------------------------------------------------------------------------
// Host-side orchestration
// ---------------------------------------------------------------------------
extern "C" void kernel_launch(void* const* d_in, const int* in_sizes, int n_in,
                              void* d_out, int out_size, void* d_ws, size_t ws_size,
                              hipStream_t stream) {
    (void)in_sizes; (void)n_in; (void)out_size; (void)ws_size;
    const float* x       = (const float*)d_in[0];
    const float* symbols = (const float*)d_in[1];
    const float* wq_attn = (const float*)d_in[2];
    const float* wk_attn = (const float*)d_in[3];
    const float* wq_rel  = (const float*)d_in[4];
    const float* wk_rel  = (const float*)d_in[5];
    const float* wr      = (const float*)d_in[6];
    const float* wv      = (const float*)d_in[7];
    const float* wo      = (const float*)d_in[8];
    const float* fcos    = (const float*)d_in[9];
    const float* fsin    = (const float*)d_in[10];

    const int M = 2 * L_SEQ;             // 4096 rows

    // bump allocator over d_ws
    size_t off = 0;
    auto alloc = [&](size_t bytes) {
        void* p = (char*)d_ws + off;
        off += (bytes + 255) & ~(size_t)255;
        return p;
    };
    bf16_t* xb   = (bf16_t*)alloc((size_t)M * DMODEL * 2);
    bf16_t* sb   = (bf16_t*)alloc((size_t)M * DMODEL * 2);
    bf16_t* wqaT = (bf16_t*)alloc((size_t)NPROJ * DMODEL * 2);
    bf16_t* wkaT = (bf16_t*)alloc((size_t)NPROJ * DMODEL * 2);
    bf16_t* wqrT = (bf16_t*)alloc((size_t)DMODEL * DMODEL * 2);
    bf16_t* wkrT = (bf16_t*)alloc((size_t)DMODEL * DMODEL * 2);
    bf16_t* wvT  = (bf16_t*)alloc((size_t)NPROJ * DMODEL * 2);
    bf16_t* woT  = (bf16_t*)alloc((size_t)NPROJ * NPROJ * 2);
    bf16_t* qhB  = (bf16_t*)alloc((size_t)M * NPROJ * 2);      // (B,8,L,64)
    bf16_t* khB  = (bf16_t*)alloc((size_t)M * NPROJ * 2);
    bf16_t* qrB  = (bf16_t*)alloc((size_t)M * DMODEL * 2);     // (B,8,L,128)
    bf16_t* krB  = (bf16_t*)alloc((size_t)M * DMODEL * 2);
    bf16_t* svTB = (bf16_t*)alloc((size_t)M * NPROJ * 2);      // (B,8,64,L)
    bf16_t* Y    = (bf16_t*)alloc((size_t)M * NPROJ * 2);

    auto cvt = [&](const float* s, bf16_t* d, int r, int c, int tr) {
        int n = r * c;
        int blocks = (n + 255) / 256;
        if (blocks > 2048) blocks = 2048;
        convert_kernel<<<blocks, 256, 0, stream>>>(s, d, r, c, tr);
    };
    cvt(x,       xb,   M,      DMODEL, 0);
    cvt(symbols, sb,   M,      DMODEL, 0);
    cvt(wq_attn, wqaT, DMODEL, NPROJ,  1);
    cvt(wk_attn, wkaT, DMODEL, NPROJ,  1);
    cvt(wq_rel,  wqrT, DMODEL, DMODEL, 1);
    cvt(wk_rel,  wkrT, DMODEL, DMODEL, 1);
    cvt(wv,      wvT,  DMODEL, NPROJ,  1);
    cvt(wo,      woT,  NPROJ,  NPROJ,  1);

    // projections (WMMA bf16), each wave owns a 16x64 tile
    gemm_bf16_kernel<<<dim3(M / 16, NPROJ / 64), 32, 0, stream>>>(
        xb, wqaT, qhB, M, NPROJ, DMODEL, 1, fcos, fsin);
    gemm_bf16_kernel<<<dim3(M / 16, NPROJ / 64), 32, 0, stream>>>(
        xb, wkaT, khB, M, NPROJ, DMODEL, 1, fcos, fsin);
    gemm_bf16_kernel<<<dim3(M / 16, DMODEL / 64), 32, 0, stream>>>(
        xb, wqrT, qrB, M, DMODEL, DMODEL, 3, nullptr, nullptr);
    gemm_bf16_kernel<<<dim3(M / 16, DMODEL / 64), 32, 0, stream>>>(
        xb, wkrT, krB, M, DMODEL, DMODEL, 3, nullptr, nullptr);
    gemm_bf16_kernel<<<dim3(M / 16, NPROJ / 64), 32, 0, stream>>>(
        sb, wvT, svTB, M, NPROJ, DMODEL, 2, nullptr, nullptr);

    // fused flash attention + rel
    attn_kernel<<<2 * (L_SEQ / 16), 256, 0, stream>>>(
        qhB, khB, qrB, krB, svTB, wr, Y);

    // final projection: out = Y @ wo  (fp32 out)
    gemm_bf16_kernel<<<dim3(M / 16, NPROJ / 64), 32, 0, stream>>>(
        Y, woT, d_out, M, NPROJ, NPROJ, 4, nullptr, nullptr);
}
